// graphtransformer_hidden_layer_mb_32212254720358
// MI455X (gfx1250) — matrix operand order repaired and checked by reading, not compiled, not run on hardware
//
#include <hip/hip_runtime.h>

typedef __attribute__((ext_vector_type(16))) _Float16 v16h;
typedef __attribute__((ext_vector_type(8)))  float    v8f;

#define IN_DIM 256
#define HEADS  8
#define DHEAD  32
#define HD     256   // HEADS * DHEAD
#define INV_SQRT_D 0.17677669529663687f  // 1/sqrt(32)
#define NEG_INF __uint_as_float(0xFF800000u)

// ---------------- 1) init: zero CSR counters ----------------
__global__ void k_init(int* __restrict__ cnt, int* __restrict__ fill, int Nd) {
    int t = blockIdx.x * blockDim.x + threadIdx.x;
    if (t < Nd) { cnt[t] = 0; fill[t] = 0; }
}

// ---------------- 2) fold a_src/a_dst into Wk/Wq (and biases) ----------------
__global__ void k_fold_attn(const float* __restrict__ Wq, const float* __restrict__ Wk,
                            const float* __restrict__ bq, const float* __restrict__ bk,
                            const float* __restrict__ a_src, const float* __restrict__ a_dst,
                            float* __restrict__ wq_att, float* __restrict__ wk_att,
                            float* __restrict__ bq_att, float* __restrict__ bk_att) {
    int t = blockIdx.x * blockDim.x + threadIdx.x;
    const int NW = HEADS * IN_DIM;
    if (t < NW) {
        int h = t / IN_DIM, i = t % IN_DIM;
        float acc = 0.0f;
        #pragma unroll
        for (int d = 0; d < DHEAD; ++d) acc += a_dst[d] * Wq[(size_t)(h * DHEAD + d) * IN_DIM + i];
        wq_att[t] = acc;
    } else if (t < 2 * NW) {
        int u = t - NW;
        int h = u / IN_DIM, i = u % IN_DIM;
        float acc = 0.0f;
        #pragma unroll
        for (int d = 0; d < DHEAD; ++d) acc += a_src[d] * Wk[(size_t)(h * DHEAD + d) * IN_DIM + i];
        wk_att[u] = acc;
    } else if (t < 2 * NW + HEADS) {
        int h = t - 2 * NW; float acc = 0.0f;
        #pragma unroll
        for (int d = 0; d < DHEAD; ++d) acc += a_dst[d] * bq[h * DHEAD + d];
        bq_att[h] = acc;
    } else if (t < 2 * NW + 2 * HEADS) {
        int h = t - 2 * NW - HEADS; float acc = 0.0f;
        #pragma unroll
        for (int d = 0; d < DHEAD; ++d) acc += a_src[d] * bk[h * DHEAD + d];
        bk_att[h] = acc;
    }
}

// ---------------- 3) per-node attention scores: feats @ w_att.T ----------------
__global__ void k_node_scores(const float* __restrict__ feats, const float* __restrict__ w_att,
                              const float* __restrict__ b_att, float* __restrict__ scores, int N) {
    int t = blockIdx.x * blockDim.x + threadIdx.x;
    if (t >= N * HEADS) return;
    int n = t >> 3, h = t & 7;
    const float* f = feats + (size_t)n * IN_DIM;
    const float* w = w_att + h * IN_DIM;
    float acc = b_att[h];
    #pragma unroll 8
    for (int i = 0; i < IN_DIM; ++i) acc += f[i] * w[i];
    scores[t] = acc;
}

// ---------------- 4) V projection GEMM via v_wmma_f32_16x16x32_f16 ----------------
__global__ void __launch_bounds__(256) k_v_gemm(const float* __restrict__ A,
                                                const float* __restrict__ Wv,
                                                const float* __restrict__ bv,
                                                float* __restrict__ V, int Ns) {
    const int lane   = threadIdx.x & 31;
    const int wave   = threadIdx.x >> 5;
    const int tile_m = blockIdx.x * 16;
    const int n0     = wave * 32;
    const int n1     = n0 + 16;
    const int colL   = lane & 15;
    const int khi    = lane >> 4;

    int ar = tile_m + colL;
    if (ar >= Ns) ar = Ns - 1;
    const float* arow  = A  + (size_t)ar * IN_DIM;
    const float* b0row = Wv + (size_t)(n0 + colL) * IN_DIM;
    const float* b1row = Wv + (size_t)(n1 + colL) * IN_DIM;

    v8f c0 = {}; v8f c1 = {};
    for (int k0 = 0; k0 < IN_DIM; k0 += 32) {
        const int ka = k0 + khi * 16;    // A: lanes 0-15 K 0..15; lanes 16-31 K 16..31
        const int kb = k0 + khi * 16;    // B: lanes 0-15 K 0..15 of col; lanes 16-31 K 16..31
        v16h a, b0, b1;
        #pragma unroll
        for (int j = 0; j < 8; ++j) {
            a[j]     = (_Float16)arow[ka + j];
            a[j + 8] = (_Float16)arow[ka + 8 + j];
        }
        #pragma unroll
        for (int j = 0; j < 16; ++j) {
            b0[j] = (_Float16)b0row[kb + j];
            b1[j] = (_Float16)b1row[kb + j];
        }
        c0 = __builtin_amdgcn_wmma_f32_16x16x32_f16(false, a, false, b0, (short)0, c0, false, false);
        c1 = __builtin_amdgcn_wmma_f32_16x16x32_f16(false, a, false, b1, (short)0, c1, false, false);
    }
    const int mbase = khi * 8;
    #pragma unroll
    for (int r = 0; r < 8; ++r) {
        int row = tile_m + mbase + r;
        if (row < Ns) {
            V[(size_t)row * HD + n0 + colL] = c0[r] + bv[n0 + colL];
            V[(size_t)row * HD + n1 + colL] = c1[r] + bv[n1 + colL];
        }
    }
}

// ---------------- 5) CSR build: count edges per dst ----------------
__global__ void k_count(const int* __restrict__ ed, int* __restrict__ cnt, int E) {
    int t = blockIdx.x * blockDim.x + threadIdx.x;
    if (t < E) atomicAdd(&cnt[ed[t]], 1);
}

// ---------------- 6) CSR build: single-block exclusive scan of counts ----------------
__global__ void __launch_bounds__(1024) k_scan(const int* __restrict__ cnt,
                                               int* __restrict__ rowptr, int Nd) {
    __shared__ int sh[1024];
    const int t = threadIdx.x;
    const int chunk = (Nd + 1023) / 1024;
    const int lo = t * chunk;
    const int hi = min(lo + chunk, Nd);
    int local = 0;
    for (int i = lo; i < hi; ++i) local += cnt[i];
    sh[t] = local;
    __syncthreads();
    // Hillis-Steele inclusive scan over 1024 partials
    for (int off = 1; off < 1024; off <<= 1) {
        int v = (t >= off) ? sh[t - off] : 0;
        __syncthreads();
        sh[t] += v;
        __syncthreads();
    }
    int run = sh[t] - local;   // exclusive prefix of this thread's chunk
    for (int i = lo; i < hi; ++i) { rowptr[i] = run; run += cnt[i]; }
    if (t == 1023) rowptr[Nd] = sh[1023];
}

// ---------------- 7) CSR build: scatter src ids into dst buckets ----------------
__global__ void k_scatter(const int* __restrict__ es, const int* __restrict__ ed,
                          const int* __restrict__ rowptr, int* __restrict__ fill,
                          int* __restrict__ esrc, int E) {
    int t = blockIdx.x * blockDim.x + threadIdx.x;
    if (t >= E) return;
    int d = ed[t];
    int pos = rowptr[d] + atomicAdd(&fill[d], 1);
    esrc[pos] = es[t];
}

// ---------------- 8) fused per-dst softmax + aggregation (atomic-free) ----------------
// One wave per dst node. Pass A: online softmax, lanes = 4 edge-slots x 8 heads.
// Pass B: lane owns 8 contiguous channels; register accumulation; plain stores.
__global__ void __launch_bounds__(256) k_dst_agg(const int* __restrict__ rowptr,
                                                 const int* __restrict__ esrc,
                                                 const float* __restrict__ ak,
                                                 const float* __restrict__ aq,
                                                 const float* __restrict__ V,
                                                 float* __restrict__ out, int Nd) {
    const int lane = threadIdx.x & 31;
    const int wave = threadIdx.x >> 5;
    const int d = blockIdx.x * 8 + wave;
    if (d >= Nd) return;
    const int start = rowptr[d];
    const int end   = rowptr[d + 1];

    // ---- Pass A: per-head max & sum(exp) via online softmax ----
    const int slot = lane >> 3;      // 0..3
    const int h    = lane & 7;       // 0..7
    const float aqh = aq[d * HEADS + h];
    float m = NEG_INF, ssum = 0.0f;
    for (int j = start + slot; j < end; j += 4) {
        int sn = esrc[j];
        float e = (ak[sn * HEADS + h] + aqh) * INV_SQRT_D;
        float mn = fmaxf(m, e);
        ssum = ssum * __expf(m - mn) + __expf(e - mn);
        m = mn;
    }
    #pragma unroll
    for (int mask = 8; mask <= 16; mask <<= 1) {   // merge the 4 slots per head
        float m2 = __shfl_xor(m, mask);
        float s2 = __shfl_xor(ssum, mask);
        float mn = fmaxf(m, m2);
        ssum = ssum * __expf(m - mn) + s2 * __expf(m2 - mn);
        m = mn;
    }
    // lane hB (0..7) now holds final (m, ssum, aqh) for head hB
    const int hB = lane >> 2;                      // head owning this lane's 8 channels
    const float mh  = __shfl(m, hB);
    const float shd = __shfl(ssum, hB);
    const float aqB = __shfl(aqh, hB);

    // ---- Pass B: weighted gather of v rows ----
    float a0 = 0, a1 = 0, a2 = 0, a3 = 0, a4 = 0, a5 = 0, a6 = 0, a7 = 0;
    const int cbase = lane * 8;
    for (int j = start; j < end; ++j) {
        int sn = esrc[j];
        float e = (ak[sn * HEADS + hB] + aqB) * INV_SQRT_D;
        float a = __expf(e - mh) / shd;
        const float4* vp = (const float4*)(V + (size_t)sn * HD + cbase);
        float4 v0 = vp[0], v1 = vp[1];
        a0 += a * v0.x; a1 += a * v0.y; a2 += a * v0.z; a3 += a * v0.w;
        a4 += a * v1.x; a5 += a * v1.y; a6 += a * v1.z; a7 += a * v1.w;
    }
    const float invd = 1.0f / fmaxf((float)(end - start), 1.0f);
    float4* op = (float4*)(out + (size_t)d * HD + cbase);
    op[0] = make_float4(a0 * invd, a1 * invd, a2 * invd, a3 * invd);
    op[1] = make_float4(a4 * invd, a5 * invd, a6 * invd, a7 * invd);
}

extern "C" void kernel_launch(void* const* d_in, const int* in_sizes, int n_in,
                              void* d_out, int out_size, void* d_ws, size_t ws_size,
                              hipStream_t stream) {
    const float* src_feats = (const float*)d_in[0];
    const float* dst_feats = (const float*)d_in[1];
    const int*   edge_src  = (const int*)d_in[2];
    const int*   edge_dst  = (const int*)d_in[3];
    const float* Wq = (const float*)d_in[4];
    const float* bq = (const float*)d_in[5];
    const float* Wk = (const float*)d_in[6];
    const float* bk = (const float*)d_in[7];
    const float* Wv = (const float*)d_in[8];
    const float* bv = (const float*)d_in[9];
    const float* a_src = (const float*)d_in[10];
    const float* a_dst = (const float*)d_in[11];

    const int Ns = in_sizes[0] / IN_DIM;
    const int Nd = in_sizes[1] / IN_DIM;
    const int E  = in_sizes[2];

    char* wsb = (char*)d_ws;
    float* V      = (float*)wsb; wsb += (size_t)Ns * HD * sizeof(float);
    float* wq_att = (float*)wsb; wsb += HEADS * IN_DIM * sizeof(float);
    float* wk_att = (float*)wsb; wsb += HEADS * IN_DIM * sizeof(float);
    float* bq_att = (float*)wsb; wsb += HEADS * sizeof(float);
    float* bk_att = (float*)wsb; wsb += HEADS * sizeof(float);
    float* attn_q = (float*)wsb; wsb += (size_t)Nd * HEADS * sizeof(float);
    float* attn_k = (float*)wsb; wsb += (size_t)Ns * HEADS * sizeof(float);
    int*   cnt    = (int*)wsb;   wsb += (size_t)Nd * sizeof(int);
    int*   fill   = (int*)wsb;   wsb += (size_t)Nd * sizeof(int);
    int*   rowptr = (int*)wsb;   wsb += ((size_t)Nd + 1) * sizeof(int);
    int*   esrc   = (int*)wsb;   wsb += (size_t)E * sizeof(int);

    float* out = (float*)d_out;
    const int B = 256;

    k_init<<<(Nd + B - 1) / B, B, 0, stream>>>(cnt, fill, Nd);
    k_fold_attn<<<(2 * HEADS * IN_DIM + 2 * HEADS + B - 1) / B, B, 0, stream>>>(
        Wq, Wk, bq, bk, a_src, a_dst, wq_att, wk_att, bq_att, bk_att);
    k_node_scores<<<((size_t)Nd * HEADS + B - 1) / B, B, 0, stream>>>(dst_feats, wq_att, bq_att, attn_q, Nd);
    k_node_scores<<<((size_t)Ns * HEADS + B - 1) / B, B, 0, stream>>>(src_feats, wk_att, bk_att, attn_k, Ns);
    k_v_gemm<<<(Ns + 15) / 16, 256, 0, stream>>>(src_feats, Wv, bv, V, Ns);
    k_count<<<(E + B - 1) / B, B, 0, stream>>>(edge_dst, cnt, E);
    k_scan<<<1, 1024, 0, stream>>>(cnt, rowptr, Nd);
    k_scatter<<<(E + B - 1) / B, B, 0, stream>>>(edge_src, edge_dst, rowptr, fill, esrc, E);
    k_dst_agg<<<(Nd + 7) / 8, 256, 0, stream>>>(rowptr, esrc, attn_k, attn_q, V, out, Nd);
}
